// ConvLayer_90280212562015
// MI455X (gfx1250) — matrix operand order repaired and checked by reading, not compiled, not run on hardware
//
#include <hip/hip_runtime.h>
#include <hip/hip_bf16.h>

typedef __attribute__((ext_vector_type(16))) _Float16 v16h;
typedef __attribute__((ext_vector_type(8)))  float    v8f;

#define NVOX 160000
#define CIN  32
#define COUT 32
#define KOFF 27
#define CONV_BLOCKS (NVOX / 128)   // 1250, exact

// ---------------------------------------------------------------------------
// Kernel 1: convert fp32 features -> f16 table (10 MB, L2-resident thereafter)
// ---------------------------------------------------------------------------
__global__ __launch_bounds__(256) void cvt_feat_kernel(
    const float* __restrict__ f, _Float16* __restrict__ o)
{
    int i = blockIdx.x * blockDim.x + threadIdx.x;   // float4 index, N*32/4 total
    float4 x = ((const float4*)f)[i];
    _Float16* d = o + 4 * i;
    d[0] = (_Float16)x.x;
    d[1] = (_Float16)x.y;
    d[2] = (_Float16)x.z;
    d[3] = (_Float16)x.w;
}

// ---------------------------------------------------------------------------
// Kernel 2: swizzle weight [27][Cin=32][Cout=32] fp32 into per-lane WMMA-B
// fragments: wB[(k*2+t)*32 + lane][16 halfs]
//   lane -> column n = (lane&15) + 16*t
//   lanes 0-15 hold K=0..15 (pair-packed per dword), lanes 16-31 hold K=16..31
// ---------------------------------------------------------------------------
__global__ __launch_bounds__(256) void swz_weight_kernel(
    const float* __restrict__ w, _Float16* __restrict__ wB)
{
    int id = blockIdx.x * blockDim.x + threadIdx.x;
    if (id >= KOFF * 2 * 32) return;
    int lane = id & 31;
    int t    = (id >> 5) & 1;
    int k    = id >> 6;
    int n     = (lane & 15) + 16 * t;
    int kbase = (lane >> 4) * 16;
    _Float16* dst = wB + id * 16;
#pragma unroll
    for (int j = 0; j < 8; ++j) {
        dst[2 * j]     = (_Float16)w[(k * CIN + (kbase + 2 * j))     * COUT + n];
        dst[2 * j + 1] = (_Float16)w[(k * CIN + (kbase + 2 * j + 1)) * COUT + n];
    }
}

// ---------------------------------------------------------------------------
// Kernel 3: gather + WMMA sparse conv. 256 threads = 8 waves; each wave does a
// 16-voxel x 32-cout tile (two 16x16 accumulators), K-loop over 27 offsets.
// Wave-uniform ballot skip: at 1.9% density most offsets contribute nothing to
// the whole 16-voxel tile, so we branch around B loads + both WMMAs (EXEC is
// all-1s at every executed WMMA since the ballot condition is wave-uniform).
// Also emits fixed-order per-block channel (sum, sumsq) partials.
// ---------------------------------------------------------------------------
__global__ __launch_bounds__(256) void conv_wmma_kernel(
    const _Float16* __restrict__ feat16,   // [N][32]
    const int*      __restrict__ nbr,      // [N][27]
    const _Float16* __restrict__ wB,       // [27*2*32][16]
    const float*    __restrict__ bias,     // [32]
    float*          __restrict__ convOut,  // [N][32]
    float*          __restrict__ partials) // [CONV_BLOCKS][64]
{
    const int lane  = threadIdx.x & 31;
    const int wave  = threadIdx.x >> 5;
    const int m     = lane & 15;           // A-matrix row this lane feeds
    const int half  = lane >> 4;           // which K-half this lane holds
    const int vbase = blockIdx.x * 128 + wave * 16;
    const int v     = vbase + m;           // voxel for this lane's A rows

    // Preload all 27 neighbor indices (issued as one load clause, latency hidden)
    int ni[KOFF];
#pragma unroll
    for (int k = 0; k < KOFF; ++k)
        ni[k] = nbr[v * KOFF + k];

    v8f c0 = {};
    v8f c1 = {};

#pragma unroll
    for (int k = 0; k < KOFF; ++k) {
        // Wave-uniform skip: no voxel in this tile has neighbor k -> zero
        // contribution, skip A/B loads and both WMMAs entirely.
        if (__builtin_amdgcn_ballot_w32(ni[k] >= 0) == 0u)
            continue;
        union { v16h h; uint4 u[2]; } A;
        if (ni[k] >= 0) {
            const uint4* frow = (const uint4*)(feat16 + ni[k] * CIN);
            A.u[0] = frow[2 * half];        // K 0..7   (lanes<16) / 16..23 (lanes>=16)
            A.u[1] = frow[2 * half + 1];    // K 8..15  (lanes<16) / 24..31 (lanes>=16)
        } else {
            A.u[0] = make_uint4(0u, 0u, 0u, 0u);
            A.u[1] = make_uint4(0u, 0u, 0u, 0u);
        }
        const v16h b0 = *(const v16h*)(wB + ((k * 2 + 0) * 32 + lane) * 16);
        const v16h b1 = *(const v16h*)(wB + ((k * 2 + 1) * 32 + lane) * 16);
        c0 = __builtin_amdgcn_wmma_f32_16x16x32_f16(false, A.h, false, b0,
                                                    (short)0, c0, false, false);
        c1 = __builtin_amdgcn_wmma_f32_16x16x32_f16(false, A.h, false, b1,
                                                    (short)0, c1, false, false);
    }

    // C/D layout: VGPR r -> row M = r + 8*half, col n = lane&15 (+16 for tile 1)
    const int n0 = lane & 15;
    const int n1 = n0 + 16;
    const float bv0 = bias[n0];
    const float bv1 = bias[n1];
    float s0 = 0.f, q0 = 0.f, s1 = 0.f, q1 = 0.f;
#pragma unroll
    for (int r = 0; r < 8; ++r) {
        const int row = vbase + r + 8 * half;
        const float y0 = c0[r] + bv0;
        const float y1 = c1[r] + bv1;
        convOut[row * COUT + n0] = y0;
        convOut[row * COUT + n1] = y1;
        s0 += y0; q0 += y0 * y0;
        s1 += y1; q1 += y1 * y1;
    }

    // Deterministic in-block reduction via LDS (fixed summation order).
    __shared__ float lsum[8][2][32];
    __shared__ float lsq [8][2][32];
    lsum[wave][0][lane] = s0;  lsq[wave][0][lane] = q0;
    lsum[wave][1][lane] = s1;  lsq[wave][1][lane] = q1;
    __syncthreads();

    const int tid = threadIdx.x;
    if (tid < 64) {
        const int kind = tid >> 5;        // 0 = sum, 1 = sumsq
        const int ch   = tid & 31;
        const int t    = ch >> 4;
        const int l0   = ch & 15;
        float acc = 0.f;
#pragma unroll
        for (int w = 0; w < 8; ++w) {
            acc += kind ? lsq[w][t][l0]      : lsum[w][t][l0];
            acc += kind ? lsq[w][t][l0 + 16] : lsum[w][t][l0 + 16];
        }
        partials[blockIdx.x * 64 + tid] = acc;
    }
}

// ---------------------------------------------------------------------------
// Kernel 4: single-block deterministic global reduction -> per-channel
// scale/shift for fused BN.
// ---------------------------------------------------------------------------
__global__ __launch_bounds__(64) void reduce_stats_kernel(
    const float* __restrict__ partials,
    const float* __restrict__ gamma,
    const float* __restrict__ beta,
    float*       __restrict__ stats)       // [0..31]=scale, [32..63]=shift
{
    __shared__ float acc[64];
    const int tid = threadIdx.x;
    float a = 0.f;
    for (int b = 0; b < CONV_BLOCKS; ++b)
        a += partials[b * 64 + tid];
    acc[tid] = a;
    __syncthreads();
    if (tid < 32) {
        const float s    = acc[tid];
        const float q    = acc[32 + tid];
        const float mean = s * (1.0f / (float)NVOX);
        const float var  = q * (1.0f / (float)NVOX) - mean * mean;
        const float inv  = rsqrtf(var + 1e-5f);
        const float scale = gamma[tid] * inv;
        stats[tid]      = scale;
        stats[32 + tid] = beta[tid] - mean * scale;
    }
}

// ---------------------------------------------------------------------------
// Kernel 5: fused BN + LeakyReLU, float4-vectorized (row = 32 ch, never splits)
// ---------------------------------------------------------------------------
__global__ __launch_bounds__(256) void bn_lrelu_kernel(
    const float* __restrict__ convOut,
    const float* __restrict__ stats,
    float*       __restrict__ out)
{
    const int i  = blockIdx.x * blockDim.x + threadIdx.x;  // float4 index
    const int c0 = (i * 4) & 31;
    float4 x = ((const float4*)convOut)[i];
    float y0 = x.x * stats[c0]     + stats[32 + c0];
    float y1 = x.y * stats[c0 + 1] + stats[33 + c0];
    float y2 = x.z * stats[c0 + 2] + stats[34 + c0];
    float y3 = x.w * stats[c0 + 3] + stats[35 + c0];
    y0 = (y0 >= 0.f) ? y0 : 0.01f * y0;
    y1 = (y1 >= 0.f) ? y1 : 0.01f * y1;
    y2 = (y2 >= 0.f) ? y2 : 0.01f * y2;
    y3 = (y3 >= 0.f) ? y3 : 0.01f * y3;
    ((float4*)out)[i] = make_float4(y0, y1, y2, y3);
}

// ---------------------------------------------------------------------------
extern "C" void kernel_launch(void* const* d_in, const int* in_sizes, int n_in,
                              void* d_out, int out_size, void* d_ws, size_t ws_size,
                              hipStream_t stream)
{
    const float* features = (const float*)d_in[0];   // [N][32] f32
    const int*   nbr      = (const int*)  d_in[1];   // [N][27] i32
    const float* weight   = (const float*)d_in[2];   // [27][32][32] f32
    const float* bias     = (const float*)d_in[3];   // [32]
    const float* gamma    = (const float*)d_in[4];   // [32]
    const float* beta     = (const float*)d_in[5];   // [32]
    float*       out      = (float*)d_out;           // [N][32] f32

    char* ws = (char*)d_ws;
    float*     convOut  = (float*)    (ws);                         // 20,480,000 B
    _Float16*  feat16   = (_Float16*) (ws + 20480000);              // 10,240,000 B
    _Float16*  wB       = (_Float16*) (ws + 30720000);              //     55,296 B
    float*     partials = (float*)    (ws + 30775296);              //    320,000 B
    float*     stats    = (float*)    (ws + 31095296);              //        256 B

    // 1) features fp32 -> f16   (N*32/4 = 1,280,000 float4 threads)
    cvt_feat_kernel<<<5000, 256, 0, stream>>>(features, feat16);
    // 2) weight swizzle into WMMA-B lane layout (27*2*32 = 1728 threads)
    swz_weight_kernel<<<7, 256, 0, stream>>>(weight, wB);
    // 3) gather + WMMA conv (+bias) with per-block stat partials, sparsity skip
    conv_wmma_kernel<<<CONV_BLOCKS, 256, 0, stream>>>(feat16, nbr, wB, bias,
                                                      convOut, partials);
    // 4) deterministic global reduction -> BN scale/shift
    reduce_stats_kernel<<<1, 64, 0, stream>>>(partials, gamma, beta, stats);
    // 5) fused BN + LeakyReLU
    bn_lrelu_kernel<<<5000, 256, 0, stream>>>(convOut, stats, out);
}
